// FF_BICONV3_58025008169024
// MI455X (gfx1250) — 8-bit matrix operand order repaired and checked by reading, compiled, not run on hardware
//
#include <hip/hip_runtime.h>
#include <hip/hip_bf16.h>

typedef __attribute__((ext_vector_type(8))) int v8i;

#define HH 256
#define WW 256
#define NN 16
#define CIN 32
#define COUT 32
#define KTOT 288
#define KPAD 320
#define TILE_W 128
#define HALO_W 130
#define XQ_OFF 16384

__global__ __launch_bounds__(256) void FF_BICONV3_prep_w(
    const float* __restrict__ wt, int* __restrict__ wq) {
  int idx = blockIdx.x * 256 + threadIdx.x;
  if (idx >= 2 * 5 * 32 * 8) return;
  int j = idx & 7;
  int r = idx >> 3;
  int lane = r & 31;
  r >>= 5;
  int c = r % 5;
  int t = r / 5;
  int cout = t * 16 + (lane & 15);
  int hi = lane >> 4;
  int kbase = c * 64 + ((j >> 1) << 4) + ((j & 1) << 2) + (hi << 3);
  int res = 0;
#pragma unroll
  for (int b = 0; b < 4; ++b) {
    int k = kbase + b;
    int sb = 0;
    if (k < KTOT) {
      int pos = k >> 5;
      int cin = k & 31;
      int kh = pos / 3;
      int kw = pos - kh * 3;
      float wv = wt[((cout * CIN + cin) * 3 + kh) * 3 + kw];
      sb = (wv >= 0.0f) ? 1 : -1;
    }
    res |= (sb & 0xFF) << (8 * b);
  }
  wq[idx] = res;
}

__global__ __launch_bounds__(256) void FF_BICONV3_prep_x(
    const float* __restrict__ x, int* __restrict__ xq) {
  int p = blockIdx.x * 256 + threadIdx.x;
  int n = p >> 16;
  int hw = p & 65535;
  const float* xp = x + (size_t)n * CIN * 65536 + hw;
#pragma unroll
  for (int c4 = 0; c4 < 8; ++c4) {
    int res = 0;
#pragma unroll
    for (int b = 0; b < 4; ++b) {
      float v = xp[(size_t)(c4 * 4 + b) * 65536];
      int s = (v > 0.0f) - (v < 0.0f);
      res |= (s & 0xFF) << (8 * b);
    }
    xq[(p << 3) + c4] = res;
  }
}

__global__ __launch_bounds__(256) void FF_BICONV3_conv(
    const int* __restrict__ xq, const int* __restrict__ wq,
    const float* __restrict__ gamma, const float* __restrict__ beta,
    const float* __restrict__ rmean, const float* __restrict__ rvar,
    float* __restrict__ out) {
  __shared__ int xs[3 * HALO_W * 8];
  __shared__ float inv_s[COUT];
  __shared__ float bias_s[COUT];

  const int tid = threadIdx.x;
  const int P0 = blockIdx.x * TILE_W;
  const int n = P0 >> 16;
  const int rem = P0 & 65535;
  const int h = rem >> 8;
  const int w0 = rem & 255;

  if (tid < COUT) {
    float iv = gamma[tid] * rsqrtf(rvar[tid] + 1e-5f);
    inv_s[tid] = iv;
    bias_s[tid] = beta[tid] - rmean[tid] * iv;
  }

  for (int j = tid; j < 3 * HALO_W * 8; j += 256) {
    int row = j / (HALO_W * 8);
    int rj = j - row * (HALO_W * 8);
    int col = rj >> 3;
    int c4 = rj & 7;
    int hh = h - 1 + row;
    int ww = w0 - 1 + col;
    int val = 0;
    if (hh >= 0 && hh < HH && ww >= 0 && ww < WW)
      val = xq[(((n * HH + hh) * WW + ww) << 3) + c4];
    xs[j] = val;
  }
  __syncthreads();

  const int lane = tid & 31;
  const int wid = tid >> 5;
  const int hi = lane >> 4;
  const int pcol = (wid << 4) + (lane & 15);

  const v8i* wqv = (const v8i*)wq;

  v8i acc0 = {0, 0, 0, 0, 0, 0, 0, 0};
  v8i acc1 = {0, 0, 0, 0, 0, 0, 0, 0};

#pragma unroll
  for (int c = 0; c < 5; ++c) {
    v8i bm = {0, 0, 0, 0, 0, 0, 0, 0};
#pragma unroll
    for (int j = 0; j < 8; ++j) {
      const int pos = c * 2 + (j >> 2);
      if (pos < 9) {
        const int kh = pos / 3;
        const int kw = pos - kh * 3;
        bm[j] = xs[((kh * HALO_W + pcol + kw) << 3) + (((j >> 1) & 1) << 2) + (hi << 1) + (j & 1)];
      }
    }
    v8i a0 = wqv[(0 * 5 + c) * 32 + lane];
    v8i a1 = wqv[(1 * 5 + c) * 32 + lane];

    acc0 = __builtin_amdgcn_wmma_i32_16x16x64_iu8(true, a0, true, bm, acc0,
                                                  false, false);
    acc1 = __builtin_amdgcn_wmma_i32_16x16x64_iu8(true, a1, true, bm, acc1,
                                                  false, false);
  }

  const int base0 = n * COUT * 65536 + h * WW + w0 + pcol;
#pragma unroll
  for (int r = 0; r < 8; ++r) {
    int m0 = (hi << 3) + r;
    float v0 = (float)acc0[r] * inv_s[m0] + bias_s[m0];
    out[base0 + m0 * 65536] = fmaxf(v0, 0.0f);
    int m1 = 16 + m0;
    float v1 = (float)acc1[r] * inv_s[m1] + bias_s[m1];
    out[base0 + m1 * 65536] = fmaxf(v1, 0.0f);
  }
}

extern "C" void kernel_launch(void* const* d_in, const int* in_sizes, int n_in,
                              void* d_out, int out_size, void* d_ws,
                              size_t ws_size, hipStream_t stream) {
  const float* x = (const float*)d_in[0];
  const float* weight = (const float*)d_in[1];
  const float* gamma = (const float*)d_in[2];
  const float* beta = (const float*)d_in[3];
  const float* rmean = (const float*)d_in[4];
  const float* rvar = (const float*)d_in[5];
  float* out = (float*)d_out;

  int* wq = (int*)d_ws;
  int* xq = (int*)((char*)d_ws + XQ_OFF);

  FF_BICONV3_prep_w<<<10, 256, 0, stream>>>(weight, wq);

  const int npix = NN * HH * WW;
  FF_BICONV3_prep_x<<<npix / 256, 256, 0, stream>>>(x, xq);

  FF_BICONV3_conv<<<npix / TILE_W, 256, 0, stream>>>(xq, wq, gamma, beta,
                                                     rmean, rvar, out);
}
